// MoEAdaptorLayerImproved_75179107549933
// MI455X (gfx1250) — compile-verified, matrix operand order checked by reading, not run on hardware
//
#include <hip/hip_runtime.h>
#include <hip/hip_bf16.h>
#include <math.h>

// ---------------------------------------------------------------------------
// MoE adaptor, fused bf16-WMMA implementation for gfx1250 (MI455X).
//   T = B*S = 16384 tokens, D_IN=768, MID=1536, D_OUT=512, E=8
// Strategy:
//   - pre-convert x / W_in / W_out / W_sc to bf16 in workspace (if it fits)
//   - fused kernel: 64-token x 512-output tile per block (8 wave32)
//     per expert: 12 MLP chunks (GEMM1 K=768 -> gelu -> LDS -> GEMM2 K=128)
//                + 6 shortcut chunks (g*x -> LDS -> GEMM2 vs W_sc)
// ---------------------------------------------------------------------------

#define T_TOK   16384
#define D_IN    768
#define MID     1536
#define D_OUT   512
#define NEXP    8

typedef __bf16 bf16_t;
typedef __attribute__((ext_vector_type(16))) __bf16 v16bf;
typedef __attribute__((ext_vector_type(8)))  __bf16 v8bf;
typedef __attribute__((ext_vector_type(8)))  float  v8f;
typedef __attribute__((ext_vector_type(4)))  float  v4f;

// LDS stride (halves) for the 64x128 staged chunk, padded vs bank conflicts
#define HLD 136

// ---------------------------------------------------------------------------
// Fragment loaders. CDNA5 16-bit A/B layout for v_wmma_*_16x16x32_bf16:
//   lane l (0..31): row/col r = l&15
//     halves[0..7]  = K[ (l>=16?8:0)      .. +7 ]
//     halves[8..15] = K[ 16+(l>=16?8:0)   .. +7 ]
// Row-major [16 x K] source => two 16-byte reads per lane (bf16 source) or
// four 16-byte reads + pack-converts (f32 source).
// ---------------------------------------------------------------------------
__device__ __forceinline__ v16bf load_frag(const bf16_t* __restrict__ base,
                                           int ld, int lane) {
    const int r  = lane & 15;
    const int hi = (lane >> 4) ? 8 : 0;
    const bf16_t* p = base + r * ld + hi;
    v8bf a0 = *(const v8bf*)(p);
    v8bf a1 = *(const v8bf*)(p + 16);
    v16bf o;
#pragma unroll
    for (int i = 0; i < 8; ++i) { o[i] = a0[i]; o[8 + i] = a1[i]; }
    return o;
}

__device__ __forceinline__ v16bf load_frag(const float* __restrict__ base,
                                           int ld, int lane) {
    const int r  = lane & 15;
    const int hi = (lane >> 4) ? 8 : 0;
    const float* p = base + r * ld + hi;
    v4f a0 = *(const v4f*)(p);
    v4f a1 = *(const v4f*)(p + 4);
    v4f a2 = *(const v4f*)(p + 16);
    v4f a3 = *(const v4f*)(p + 20);
    v16bf o;
#pragma unroll
    for (int i = 0; i < 4; ++i) {
        o[i]      = (__bf16)a0[i];
        o[4 + i]  = (__bf16)a1[i];
        o[8 + i]  = (__bf16)a2[i];
        o[12 + i] = (__bf16)a3[i];
    }
    return o;
}

__device__ __forceinline__ v16bf load_frag_lds(const bf16_t* base, int ld, int lane) {
    const int r  = lane & 15;
    const int hi = (lane >> 4) ? 8 : 0;
    const bf16_t* p = base + r * ld + hi;
    v8bf a0 = *(const v8bf*)(p);
    v8bf a1 = *(const v8bf*)(p + 16);
    v16bf o;
#pragma unroll
    for (int i = 0; i < 8; ++i) { o[i] = a0[i]; o[8 + i] = a1[i]; }
    return o;
}

__device__ __forceinline__ v8f wmma_bf16(v16bf a, v16bf b, v8f c) {
    return __builtin_amdgcn_wmma_f32_16x16x32_bf16(false, a, false, b,
                                                   (short)0, c, false, false);
}

__device__ __forceinline__ float gelu_exact(float x) {
    return 0.5f * x * (1.0f + erff(x * 0.70710678118654752f));
}

// ---------------------------------------------------------------------------
// Kernel: f32 -> bf16 conversion (8 elements / thread), n % 2048 == 0 here
// ---------------------------------------------------------------------------
__global__ void cvt_bf16_kernel(const float* __restrict__ src,
                                bf16_t* __restrict__ dst, int n) {
    int i = (blockIdx.x * blockDim.x + threadIdx.x) * 8;
    if (i >= n) return;
    v4f a = *(const v4f*)(src + i);
    v4f b = *(const v4f*)(src + i + 4);
    v8bf o;
#pragma unroll
    for (int k = 0; k < 4; ++k) { o[k] = (__bf16)a[k]; o[4 + k] = (__bf16)b[k]; }
    *(v8bf*)(dst + i) = o;
}

// ---------------------------------------------------------------------------
// Kernel: gating softmax. One thread per (token, expert); 8 adjacent lanes
// per token cooperate via wave32 shuffles.
// ---------------------------------------------------------------------------
__global__ void moe_gates_kernel(const float* __restrict__ x,
                                 const float* __restrict__ w_gate,
                                 float* __restrict__ gates) {
    int gid = blockIdx.x * blockDim.x + threadIdx.x;   // [0, T*E)
    int t = gid >> 3;
    int e = gid & 7;
    const float* xr = x + (size_t)t * D_IN;
    float s = 0.0f;
#pragma unroll 8
    for (int d = 0; d < D_IN; ++d) s += xr[d] * w_gate[d * NEXP + e];
    float m = s;
#pragma unroll
    for (int off = 1; off < 8; off <<= 1) m = fmaxf(m, __shfl_xor(m, off, 32));
    float ex = __expf(s - m);
    float sum = ex;
#pragma unroll
    for (int off = 1; off < 8; off <<= 1) sum += __shfl_xor(sum, off, 32);
    gates[gid] = ex / sum;
}

// ---------------------------------------------------------------------------
// Kernel: per-expert constant c[e,m] = bias_in[e,:] . W_in[e,m,:]
// ---------------------------------------------------------------------------
__global__ void moe_cvals_kernel(const float* __restrict__ bias_in,
                                 const float* __restrict__ W_in,
                                 float* __restrict__ cvals) {
    int gid = blockIdx.x * blockDim.x + threadIdx.x;   // [0, E*MID)
    int e = gid / MID;
    int m = gid - e * MID;
    const float* b = bias_in + (size_t)e * D_IN;
    const float* w = W_in + ((size_t)e * MID + m) * D_IN;
    float s = 0.0f;
#pragma unroll 8
    for (int d = 0; d < D_IN; ++d) s += b[d] * w[d];
    cvals[gid] = s;
}

// ---------------------------------------------------------------------------
// Fused kernel (templated on matrix-source type: bf16 fast path, f32 fallback)
// ---------------------------------------------------------------------------
template <typename SrcT>
__global__ void __launch_bounds__(256)
moe_fused_kernel(const SrcT* __restrict__ x,
                 const float* __restrict__ gates,
                 const float* __restrict__ cvals,
                 const SrcT* __restrict__ W_in,
                 const SrcT* __restrict__ W_out,
                 const float* __restrict__ b_out,
                 const SrcT* __restrict__ W_sc,
                 float* __restrict__ out) {
    __shared__ float  gate_s[64 * NEXP];          // 2 KB
    __shared__ bf16_t Hs[64 * HLD];               // 17.4 KB

    const int lane = threadIdx.x & 31;
    const int wave = threadIdx.x >> 5;
    const int t0   = blockIdx.x * 64;             // token tile base
    const int n0   = wave * 64;                   // output-column base (wave)
    const int r    = lane & 15;
    const int rhi  = (lane >> 4) ? 8 : 0;

    for (int i = threadIdx.x; i < 64 * NEXP; i += 256)
        gate_s[i] = gates[(size_t)t0 * NEXP + i];
    __syncthreads();

    v8f acc[4][4];
#pragma unroll
    for (int fm = 0; fm < 4; ++fm)
#pragma unroll
        for (int fn = 0; fn < 4; ++fn) acc[fm][fn] = (v8f)0.0f;

    // GEMM1 decomposition: 8 waves tile the 64x128 H chunk as 4x2 of (16x64)
    const int hm0 = (wave & 3) * 16;
    const int hn0 = (wave >> 2) * 64;

    // x-copy decomposition for shortcut chunks: 16 (rows) x 16 (col-octets)
    const int crow = threadIdx.x >> 4;            // 0..15
    const int ccol = (threadIdx.x & 15) * 8;      // 0..120

    for (int e = 0; e < NEXP; ++e) {
        const SrcT* Win_e  = W_in  + (size_t)e * MID   * D_IN;
        const SrcT* Wout_e = W_out + (size_t)e * D_OUT * MID;
        const SrcT* Wsc_e  = W_sc  + (size_t)e * D_OUT * D_IN;

        // 12 MLP chunks + 6 shortcut chunks, all through the same LDS tile.
        for (int chunk = 0; chunk < 18; ++chunk) {
            const SrcT* Bbase;
            int bld;
            if (chunk < 12) {
                const int mc = chunk * 128;
                // ---- GEMM1: H rows [hm0,hm0+16) x cols [hn0,hn0+64) ----
                v8f h[4];
#pragma unroll
                for (int fn = 0; fn < 4; ++fn) h[fn] = (v8f)0.0f;
                for (int k = 0; k < D_IN; k += 32) {
                    v16bf a = load_frag(x + (size_t)(t0 + hm0) * D_IN + k,
                                        D_IN, lane);
#pragma unroll
                    for (int fn = 0; fn < 4; ++fn) {
                        v16bf b = load_frag(
                            Win_e + (size_t)(mc + hn0 + fn * 16) * D_IN + k,
                            D_IN, lane);
                        h[fn] = wmma_bf16(a, b, h[fn]);
                    }
                }
                __syncthreads();   // previous chunk's GEMM2 reads done
                // epilogue: g * gelu(h - c) -> LDS bf16
#pragma unroll
                for (int fn = 0; fn < 4; ++fn) {
                    int col = hn0 + fn * 16 + r;
                    float cv = cvals[e * MID + mc + col];
#pragma unroll
                    for (int v = 0; v < 8; ++v) {
                        int row = hm0 + rhi + v;
                        float g = gate_s[row * NEXP + e];
                        Hs[row * HLD + col] =
                            (__bf16)(g * gelu_exact(h[fn][v] - cv));
                    }
                }
                Bbase = Wout_e + mc;
                bld   = MID;
            } else {
                const int kc = (chunk - 12) * 128;
                __syncthreads();   // previous chunk's GEMM2 reads done
                // ---- copy g * x[:, kc..kc+128) -> LDS bf16 ----
#pragma unroll
                for (int it = 0; it < 4; ++it) {
                    int row = crow + it * 16;
                    float g = gate_s[row * NEXP + e];
                    const SrcT* src = x + (size_t)(t0 + row) * D_IN + kc + ccol;
                    bf16_t* dst = &Hs[row * HLD + ccol];
#pragma unroll
                    for (int i = 0; i < 8; ++i)
                        dst[i] = (__bf16)(g * (float)src[i]);
                }
                Bbase = Wsc_e + kc;
                bld   = D_IN;
            }
            __syncthreads();
            // ---- GEMM2: acc += Hs(64x128) @ B_chunk^T ----
            for (int k = 0; k < 128; k += 32) {
                v16bf a[4];
#pragma unroll
                for (int fm = 0; fm < 4; ++fm)
                    a[fm] = load_frag_lds(Hs + (fm * 16) * HLD + k, HLD, lane);
#pragma unroll
                for (int fn = 0; fn < 4; ++fn) {
                    v16bf b = load_frag(
                        Bbase + (size_t)(n0 + fn * 16) * bld + k, bld, lane);
#pragma unroll
                    for (int fm = 0; fm < 4; ++fm)
                        acc[fm][fn] = wmma_bf16(a[fm], b, acc[fm][fn]);
                }
            }
        }

        // ---- gated bias: acc += g * b_out[e, n] ----
#pragma unroll
        for (int fn = 0; fn < 4; ++fn) {
            float bo = b_out[e * D_OUT + n0 + fn * 16 + r];
#pragma unroll
            for (int fm = 0; fm < 4; ++fm) {
#pragma unroll
                for (int v = 0; v < 8; ++v) {
                    float g = gate_s[(fm * 16 + rhi + v) * NEXP + e];
                    acc[fm][fn][v] += g * bo;
                }
            }
        }
    }

    // ---- store: lane holds column n, rows fm*16 + rhi + v ----
#pragma unroll
    for (int fm = 0; fm < 4; ++fm) {
#pragma unroll
        for (int fn = 0; fn < 4; ++fn) {
            int nn = n0 + fn * 16 + r;
#pragma unroll
            for (int v = 0; v < 8; ++v) {
                int row = t0 + fm * 16 + rhi + v;
                out[(size_t)row * D_OUT + nn] = acc[fm][fn][v];
            }
        }
    }
}

// ---------------------------------------------------------------------------
extern "C" void kernel_launch(void* const* d_in, const int* in_sizes, int n_in,
                              void* d_out, int out_size, void* d_ws, size_t ws_size,
                              hipStream_t stream) {
    (void)in_sizes; (void)n_in; (void)out_size;
    const float* x       = (const float*)d_in[0];
    const float* w_gate  = (const float*)d_in[1];
    const float* bias_in = (const float*)d_in[2];
    const float* W_in    = (const float*)d_in[3];
    const float* W_out   = (const float*)d_in[4];
    const float* b_out   = (const float*)d_in[5];
    const float* W_sc    = (const float*)d_in[6];
    float* out = (float*)d_out;

    const size_t N_X    = (size_t)T_TOK * D_IN;          // 12,582,912
    const size_t N_WIN  = (size_t)NEXP * MID * D_IN;     //  9,437,184
    const size_t N_WOUT = (size_t)NEXP * D_OUT * MID;    //  6,291,456
    const size_t N_WSC  = (size_t)NEXP * D_OUT * D_IN;   //  3,145,728

    float* gates = (float*)d_ws;                         // T*E f32 = 512 KB
    float* cvals = gates + (size_t)T_TOK * NEXP;         // E*MID f32 = 48 KB
    bf16_t* xb    = (bf16_t*)(cvals + (size_t)NEXP * MID);
    bf16_t* Winb  = xb + N_X;
    bf16_t* Woutb = Winb + N_WIN;
    bf16_t* Wscb  = Woutb + N_WOUT;

    const size_t need = (size_t)(T_TOK * NEXP + NEXP * MID) * sizeof(float)
                      + (N_X + N_WIN + N_WOUT + N_WSC) * sizeof(bf16_t);

    moe_gates_kernel<<<(T_TOK * NEXP) / 256, 256, 0, stream>>>(x, w_gate, gates);
    moe_cvals_kernel<<<(NEXP * MID) / 256, 256, 0, stream>>>(bias_in, W_in, cvals);

    if (ws_size >= need) {
        // Fast path: bf16 weights/activations staged in workspace.
        cvt_bf16_kernel<<<N_X    / 2048, 256, 0, stream>>>(x,     xb,    (int)N_X);
        cvt_bf16_kernel<<<N_WIN  / 2048, 256, 0, stream>>>(W_in,  Winb,  (int)N_WIN);
        cvt_bf16_kernel<<<N_WOUT / 2048, 256, 0, stream>>>(W_out, Woutb, (int)N_WOUT);
        cvt_bf16_kernel<<<N_WSC  / 2048, 256, 0, stream>>>(W_sc,  Wscb,  (int)N_WSC);
        moe_fused_kernel<bf16_t><<<T_TOK / 64, 256, 0, stream>>>(
            xb, gates, cvals, Winb, Woutb, b_out, Wscb, out);
    } else {
        // Fallback: convert f32 -> bf16 in-register inside the hot loop.
        moe_fused_kernel<float><<<T_TOK / 64, 256, 0, stream>>>(
            x, gates, cvals, W_in, W_out, b_out, W_sc, out);
    }
}
